// Head_31885837205638
// MI455X (gfx1250) — matrix operand order repaired and checked by reading, not compiled, not run on hardware
//
#include <hip/hip_runtime.h>
#include <hip/hip_bf16.h>

// ---------------------------------------------------------------------------
// Causal attention head for MI455X (gfx1250), bf16 WMMA + TDM async tiles.
//   S=4096, B=2, D_MODEL=1024, D_FF=256
// Kernel 0: weight transpose+convert: W[1024][256] fp32 -> Wt[256][1024] bf16
// Kernel 1: QKV projection (x fp32 -> bf16): Q,K as [B][S][256], V transposed
//           as Vt[B][256][S]. 4 n-tiles per block to reuse x A-fragments.
// Kernel 2: flash attention, one wave per (batch, 16-query tile); K/V tiles
//           are DMA'd into LDS by the Tensor Data Mover, double-buffered.
// ---------------------------------------------------------------------------

#define S_LEN   4096
#define BATCH   2
#define DMODEL  1024
#define DFF     256

typedef __bf16 bf16;
typedef __attribute__((ext_vector_type(16))) __bf16 v16bf;
typedef __attribute__((ext_vector_type(8)))  float  v8f;
typedef __attribute__((ext_vector_type(4)))  unsigned int u32x4;
typedef __attribute__((ext_vector_type(8)))  unsigned int u32x8;

// Fragment coordinate maps (wave32, v_wmma_f32_16x16x32_bf16), CDNA5 ISA 7.12.2:
//  A (16x32, MxK): M = lane&15;  element e -> K = e + 16*(lane>=16) (contiguous order used here)
//  B (32x16, KxN): N = lane&15;  element e -> K = e + 16*(lane>=16) (this code's order, same as A)
//  C/D (16x16):    N = lane&15;  element r -> M = r + 8*(lane>=16)

__device__ __forceinline__ float half_max(float v) {
#pragma unroll
  for (int off = 1; off < 16; off <<= 1)
    v = fmaxf(v, __shfl_xor(v, off, 32));
  return v;
}
__device__ __forceinline__ float half_sum(float v) {
#pragma unroll
  for (int off = 1; off < 16; off <<= 1)
    v += __shfl_xor(v, off, 32);
  return v;
}

// ---- Tensor Data Mover: 2D tile (2-byte elements) global -> LDS -----------
// D# layout per CDNA5 ISA 8.3/8.4 (group0: 4 SGPRs, group1: 8 SGPRs).
__device__ __forceinline__ void tdm_load_2d(unsigned lds_off, const void* gptr,
                                            unsigned tensor_d0, unsigned tensor_d1,
                                            unsigned tile_d0, unsigned tile_d1,
                                            unsigned stride_d0) {
  unsigned long long ga = (unsigned long long)(uintptr_t)gptr;
  u32x4 g0;
  g0[0] = 1u;                                   // count=1, user descriptor
  g0[1] = lds_off;                              // lds_addr (bytes)
  g0[2] = (unsigned)(ga & 0xffffffffu);         // global_addr[31:0]
  g0[3] = (unsigned)((ga >> 32) & 0x01ffffffu)  // global_addr[56:32]
          | (2u << 30);                         // type = 2 ("image")
  u32x8 g1;
  g1[0] = (1u << 16);                           // data_size=1 -> 2 bytes
  g1[1] = (tensor_d0 & 0xffffu) << 16;          // tensor_dim0[15:0]
  g1[2] = (tensor_d0 >> 16) | ((tensor_d1 & 0xffffu) << 16);
  g1[3] = (tensor_d1 >> 16) | ((tile_d0 & 0xffffu) << 16);
  g1[4] = (tile_d1 & 0xffffu);                  // tile_dim1 (tile_dim2 = 0)
  g1[5] = stride_d0;                            // tensor_dim0_stride[31:0]
  g1[6] = 0;                                    // stride hi, dim1_stride lo
  g1[7] = 0;
  asm volatile("tensor_load_to_lds %0, %1" :: "s"(g0), "s"(g1) : "memory");
}

// ---------------- Kernel 0: weight transpose + fp32->bf16 ------------------
// grid.x = DMODEL*DFF/256, grid.y = 3; block = 256.
__global__ __launch_bounds__(256) void wtr_kernel(
    const float* __restrict__ Wq, const float* __restrict__ Wk,
    const float* __restrict__ Wv, bf16* __restrict__ Wt) {
  const int w = blockIdx.y;
  const float* W = (w == 0) ? Wq : ((w == 1) ? Wk : Wv);
  bf16* T = Wt + (size_t)w * DFF * DMODEL;
  const int idx = blockIdx.x * 256 + threadIdx.x;  // coalesced read
  const int k = idx >> 8;        // row in W (d_model)
  const int f = idx & (DFF - 1); // col in W (d_ff)
  T[(size_t)f * DMODEL + k] = (bf16)W[idx];
}

// ------------------------- QKV projection ----------------------------------
// grid.x = (8192/16) * (256/64) = 2048, grid.y = 3; block = 32 (1 wave).
// Each wave: one 16-row m-tile x four 16-wide n-tiles (A-fragment reuse x4).
__global__ __launch_bounds__(32) void qkv_kernel(
    const float* __restrict__ x, const bf16* __restrict__ Wt,
    bf16* __restrict__ Qb, bf16* __restrict__ Kb, bf16* __restrict__ Vt) {
  const int lane  = threadIdx.x;
  const int laneN = lane & 15;
  const int hiL   = lane >> 4;

  const int mt = blockIdx.x >> 2;       // 0..511 (16-row tiles of x)
  const int ng = blockIdx.x & 3;        // 0..3   (64-feature group)

  const int w = blockIdx.y;
  const bf16* Wg = Wt + (size_t)w * DFF * DMODEL;
  bf16*      Dst = (w == 0) ? Qb : ((w == 1) ? Kb : Vt);

  v8f acc[4];
#pragma unroll
  for (int t = 0; t < 4; ++t) acc[t] = (v8f){};

  const int    row = mt * 16 + laneN;   // A-fragment row (M = lane&15)
  const float* xr  = x + (size_t)row * DMODEL;

  for (int kc = 0; kc < DMODEL / 32; ++kc) {
    // A fragment: x row fp32 -> bf16 (one contiguous 16-float run per lane)
    v16bf a;
    const float* xq = xr + kc * 32 + 16 * hiL;
#pragma unroll
    for (int i = 0; i < 8; ++i) {
      a[i]     = (bf16)xq[i];
      a[i + 8] = (bf16)xq[i + 8];
    }
    // 4 B fragments: contiguous 32B v16bf loads from transposed weights
#pragma unroll
    for (int t = 0; t < 4; ++t) {
      const int f = ng * 64 + t * 16 + laneN;      // B-fragment column (N)
      const bf16* wp = Wg + (size_t)f * DMODEL + kc * 32 + 16 * hiL;
      v16bf bfr = *(const v16bf*)wp;
      acc[t] = __builtin_amdgcn_wmma_f32_16x16x32_bf16(
          false, a, false, bfr, (short)0, acc[t], false, false);
    }
  }

  // Store: flat row m = s*B + b. Q,K -> [b][s][f]; V -> transposed [b][f][s].
  // Single wave-uniform branch; unrolled store loops are branch-free.
  if (w == 2) {
#pragma unroll
    for (int t = 0; t < 4; ++t)
#pragma unroll
      for (int r = 0; r < 8; ++r) {
        const int m = mt * 16 + r + 8 * hiL;
        const int f = ng * 64 + t * 16 + laneN;
        Dst[((size_t)(m & 1) * DFF + f) * S_LEN + (m >> 1)] = (bf16)acc[t][r];
      }
  } else {
#pragma unroll
    for (int t = 0; t < 4; ++t)
#pragma unroll
      for (int r = 0; r < 8; ++r) {
        const int m = mt * 16 + r + 8 * hiL;
        const int f = ng * 64 + t * 16 + laneN;
        Dst[((size_t)(m & 1) * S_LEN + (m >> 1)) * DFF + f] = (bf16)acc[t][r];
      }
  }
}

// ------------------------- Flash attention ---------------------------------
// grid.x = S/16 query tiles, grid.y = batch, block = 32 (one wave).
__global__ __launch_bounds__(32) void attn_kernel(
    const bf16* __restrict__ Qb, const bf16* __restrict__ Kb,
    const bf16* __restrict__ Vt, float* __restrict__ out) {
  __shared__ bf16 Kls[2][32 * DFF];   // K tile  [k=32][f=256], 16 KB each
  __shared__ bf16 Vls[2][DFF * 32];   // Vt tile [f=256][k=32], 16 KB each
  __shared__ bf16 Pt[16 * 32];        // P tile staging (C/D -> A layout)

  const int lane  = threadIdx.x;
  const int laneN = lane & 15;
  const int hiL   = lane >> 4;

  const int qbase = blockIdx.x * 16;
  const int b     = blockIdx.y;

  const bf16* Q = Qb + (size_t)b * S_LEN * DFF;
  const bf16* K = Kb + (size_t)b * S_LEN * DFF;
  const bf16* V = Vt + (size_t)b * DFF * S_LEN;   // transposed [f][s]

  // Preload the whole Q tile as 8 A-fragments (16 x 256 bf16).
  v16bf qf[8];
  {
    const bf16* qr = Q + (size_t)(qbase + laneN) * DFF;
#pragma unroll
    for (int fc = 0; fc < 8; ++fc) {
      const bf16* p = qr + fc * 32 + 16 * hiL;
      v16bf a;
#pragma unroll
      for (int i = 0; i < 8; ++i) { a[i] = p[i]; a[i + 8] = p[i + 8]; }
      qf[fc] = a;
    }
  }

  v8f O[16];
#pragma unroll
  for (int t = 0; t < 16; ++t) O[t] = (v8f){};

  float m_r[8], l_r[8];
#pragma unroll
  for (int r = 0; r < 8; ++r) { m_r[r] = -1e30f; l_r[r] = 0.0f; }

  const float scale = 0.0625f;                  // 1/sqrt(256)
  const int   nkt   = (qbase + 16 + 31) >> 5;   // causal 32-wide k-tiles

  // TDM: prime buffer 0 (K tile: 32 rows x 256 contig; V tile: 256 rows x 32)
  tdm_load_2d((unsigned)(uintptr_t)&Kls[0][0], K, DFF, 32, DFF, 32, DFF);
  tdm_load_2d((unsigned)(uintptr_t)&Vls[0][0], V, S_LEN, DFF, 32, DFF, S_LEN);

  for (int kt = 0; kt < nkt; ++kt) {
    const int kbase = kt * 32;
    const int cur   = kt & 1;

    if (kt + 1 < nkt) {   // prefetch next pair into the other buffer
      const int nb = kbase + 32;
      tdm_load_2d((unsigned)(uintptr_t)&Kls[cur ^ 1][0],
                  K + (size_t)nb * DFF, DFF, 32, DFF, 32, DFF);
      tdm_load_2d((unsigned)(uintptr_t)&Vls[cur ^ 1][0],
                  V + nb, S_LEN, DFF, 32, DFF, S_LEN);
      __builtin_amdgcn_s_wait_tensorcnt(2);   // current pair done, next in flight
    } else {
      __builtin_amdgcn_s_wait_tensorcnt(0);
    }
    asm volatile("" ::: "memory");

    // ---- scores: S[16q x 32k] = Q . K^T (B-frags: contiguous LDS v16bf) ----
    v8f sc0 = {}, sc1 = {};
#pragma unroll
    for (int fc = 0; fc < 8; ++fc) {
      const bf16* kp0 = &Kls[cur][(size_t)laneN * DFF + fc * 32 + 16 * hiL];
      const bf16* kp1 = kp0 + (size_t)16 * DFF;
      v16bf kb0 = *(const v16bf*)kp0;
      v16bf kb1 = *(const v16bf*)kp1;
      sc0 = __builtin_amdgcn_wmma_f32_16x16x32_bf16(
          false, qf[fc], false, kb0, (short)0, sc0, false, false);
      sc1 = __builtin_amdgcn_wmma_f32_16x16x32_bf16(
          false, qf[fc], false, kb1, (short)0, sc1, false, false);
    }

    // ---- scale, causal mask, online softmax ----
    float alpha[8];
#pragma unroll
    for (int r = 0; r < 8; ++r) {
      const int q  = qbase + r + 8 * hiL;
      const int k0 = kbase + laneN;
      const int k1 = kbase + 16 + laneN;
      float s0 = sc0[r] * scale; if (k0 > q) s0 = -1e30f;
      float s1 = sc1[r] * scale; if (k1 > q) s1 = -1e30f;

      float mn = half_max(fmaxf(s0, s1));
      mn = fmaxf(m_r[r], mn);
      const float a  = __expf(m_r[r] - mn);
      const float p0 = __expf(s0 - mn);
      const float p1 = __expf(s1 - mn);
      l_r[r] = l_r[r] * a + half_sum(p0 + p1);
      m_r[r] = mn;
      alpha[r] = a;

      const int M = r + 8 * hiL;
      Pt[M * 32 + laneN]      = (bf16)p0;
      Pt[M * 32 + 16 + laneN] = (bf16)p1;
    }

#pragma unroll
    for (int t = 0; t < 16; ++t)
#pragma unroll
      for (int r = 0; r < 8; ++r) O[t][r] *= alpha[r];

    __syncthreads();   // single wave; LDS RAW ordering via DScnt

    v16bf pa;
    {
      const bf16* pr = &Pt[laneN * 32 + 16 * hiL];
#pragma unroll
      for (int i = 0; i < 8; ++i) { pa[i] = pr[i]; pa[i + 8] = pr[i + 8]; }
    }

    // ---- O += P . V : B-frags are contiguous LDS v16bf from Vt tile ----
#pragma unroll
    for (int t = 0; t < 16; ++t) {
      const bf16* vp = &Vls[cur][(size_t)(t * 16 + laneN) * 32 + 16 * hiL];
      v16bf vb = *(const v16bf*)vp;
      O[t] = __builtin_amdgcn_wmma_f32_16x16x32_bf16(
          false, pa, false, vb, (short)0, O[t], false, false);
    }
    __syncthreads();
  }

  // ---- epilogue: divide by softmax denominator, store fp32 [S,B,256] ----
#pragma unroll
  for (int t = 0; t < 16; ++t) {
#pragma unroll
    for (int r = 0; r < 8; ++r) {
      const int s = qbase + r + 8 * hiL;
      out[((size_t)s * BATCH + b) * DFF + t * 16 + laneN] = O[t][r] / l_r[r];
    }
  }
}

// ---------------------------------------------------------------------------
extern "C" void kernel_launch(void* const* d_in, const int* in_sizes, int n_in,
                              void* d_out, int out_size, void* d_ws,
                              size_t ws_size, hipStream_t stream) {
  const float* x  = (const float*)d_in[0];
  // d_in[1] = additive causal mask: computed analytically, not read.
  const float* Wq = (const float*)d_in[2];
  const float* Wk = (const float*)d_in[3];
  const float* Wv = (const float*)d_in[4];
  float* out = (float*)d_out;

  const size_t per = (size_t)BATCH * S_LEN * DFF;   // 2 Mi elements
  bf16* Qb = (bf16*)d_ws;
  bf16* Kb = Qb + per;
  bf16* Vt = Kb + per;
  bf16* Wt = Vt + per;   // 3 x [256][1024] bf16 = 1.5 MiB (ws total 13.5 MiB)

  dim3 g0(DMODEL * DFF / 256, 3);
  wtr_kernel<<<g0, 256, 0, stream>>>(Wq, Wk, Wv, Wt);

  dim3 g1((S_LEN * BATCH / 16) * (DFF / 64), 3);     // 2048 x 3
  qkv_kernel<<<g1, 32, 0, stream>>>(x, Wt, Qb, Kb, Vt);

  dim3 g2(S_LEN / 16, BATCH);                        // 256 x 2 query tiles
  attn_kernel<<<g2, 32, 0, stream>>>(Qb, Kb, Vt, out);
}
